// QuantAttentionFused_24017457119256
// MI455X (gfx1250) — matrix operand order repaired and checked by reading, not compiled, not run on hardware
//
#include <hip/hip_runtime.h>
#include <hip/hip_bf16.h>

#define SLEN   2048
#define HDIM   4096
#define NHEADS 32
#define NKVH   8
#define HEADD  128
#define QKV_N  ((NHEADS + 2 * NKVH) * HEADD)   /* 6144 */

typedef __attribute__((ext_vector_type(16))) __bf16 v16bf;
typedef __attribute__((ext_vector_type(8)))  __bf16 v8bf;
typedef __attribute__((ext_vector_type(8)))  float  v8f;

// ---------------------------------------------------------------------------
// Tensor Data Mover support (toolchain-dependent builtin arity; full fallback)
// ---------------------------------------------------------------------------
#if defined(__has_builtin)
#  if __has_builtin(__builtin_amdgcn_tensor_load_to_lds) && \
      __has_builtin(__builtin_amdgcn_s_wait_tensorcnt)
#    define USE_TDM 1
#  endif
#endif
#ifndef USE_TDM
#  define USE_TDM 0
#endif

static __device__ __forceinline__ unsigned lds_off_u32(const void* p) {
  // generic LDS pointer: addr[63:32]=SHARED_BASE aperture, addr[31:0]=LDS byte
  return (unsigned)(unsigned long long)p;
}

#if USE_TDM
typedef __attribute__((ext_vector_type(4))) unsigned tdm_v4u;
typedef __attribute__((ext_vector_type(8))) int      tdm_v8i;
typedef __attribute__((ext_vector_type(4))) int      tdm_v4i;

// 2D tile DMA: LDS[lds_addr ...] <= tile_d1 rows x tile_d0 elements,
// row stride d0_stride (elements), element size 1<<ds_log2 bytes.
static __device__ __forceinline__ void tdm_load_2d(
    unsigned lds_byte_addr, const void* gptr, unsigned ds_log2,
    unsigned tile_d0, unsigned tile_d1, unsigned tensor_d0, unsigned tensor_d1,
    unsigned long long d0_stride) {
  const unsigned long long ga = (unsigned long long)gptr;
  tdm_v4u g0;
  g0[0] = 1u;                                              // count=1, user D#
  g0[1] = lds_byte_addr;                                   // lds_addr
  g0[2] = (unsigned)(ga & 0xFFFFFFFFu);                    // global_addr lo
  g0[3] = (unsigned)((ga >> 32) & 0x01FFFFFFu) | (2u << 30);  // hi + type=2
  tdm_v8i g1;
  g1[0] = (int)(ds_log2 << 16);                            // wg_mask=0, flags=0
  g1[1] = (int)((tensor_d0 & 0xFFFFu) << 16);              // dim0[15:0]
  g1[2] = (int)((tensor_d0 >> 16) | ((tensor_d1 & 0xFFFFu) << 16));
  g1[3] = (int)((tensor_d1 >> 16) | (tile_d0 << 16));
  g1[4] = (int)(tile_d1 & 0xFFFFu);                        // tile_dim2 = 0
  g1[5] = (int)(d0_stride & 0xFFFFFFFFull);
  g1[6] = (int)((d0_stride >> 32) & 0xFFFFull);            // dim1_stride = 0
  g1[7] = 0;
  tdm_v4i g2 = {1, 0, 0, 0};                               // tensor_dim2=1
  tdm_v4i g3 = {0, 0, 0, 0};
#  if __has_include(<hip/amd_detail/amd_gfx1250_TDM.h>)
  tdm_v8i g23 = {1, 0, 0, 0, 0, 0, 0, 0};                  // clang-23 6-arg form
  __builtin_amdgcn_tensor_load_to_lds(g0, g1, g2, g3, g23, 0);
#  else
  __builtin_amdgcn_tensor_load_to_lds(g0, g1, g2, g3, 0);  // ROCm 7.2 5-arg form
#  endif
}
#endif

static __device__ __forceinline__ unsigned short f2bf(float f) {
  union { float f; unsigned u; } x; x.f = f;
  unsigned r = x.u + 0x7FFFu + ((x.u >> 16) & 1u);   // round-to-nearest-even
  return (unsigned short)(r >> 16);
}

union AFragU { v16bf v; v8bf h[2]; unsigned short u[16]; };

// A-fragment: elements 0..7 at p0[0..7], elements 8..15 at p1[0..7]
static __device__ __forceinline__ v16bf load_frag_2x8(const unsigned short* p0,
                                                      const unsigned short* p1) {
  AFragU f;
  f.h[0] = *(const v8bf*)p0;
  f.h[1] = *(const v8bf*)p1;
  return f.v;
}

static __device__ __forceinline__ v8f vzero8() {
  v8f z = {0.f, 0.f, 0.f, 0.f, 0.f, 0.f, 0.f, 0.f};
  return z;
}

// ---------------------------------------------------------------------------
// bf16-WMMA GEMM: C[M,N] fp32 = A[M,K] x B[K,N].  A fp32 (converted while
// staging) or bf16 (ABF16). B fp32, staged by TDM then converted in LDS.
// 128 threads (4 waves), WG tile 64x64, K-step 32.
// ---------------------------------------------------------------------------
template <bool ABF16>
__global__ __launch_bounds__(128) void gemm_bf16_kernel(
    const void* __restrict__ Aptr, const float* __restrict__ B,
    float* __restrict__ C, int M, int N, int K) {
  __shared__ alignas(32) unsigned short lA[64 * 32];   // A tile  [row][k]
  __shared__ alignas(32) unsigned short lBt[64 * 32];  // B tile  [n][k]
  __shared__ alignas(16) float          lBf[32 * 64];  // raw fp32 B tile [k][n]

  const int tid  = threadIdx.x;
  const int lane = tid & 31;
  const int wave = tid >> 5;
  const int l16  = lane & 15;
  const int g    = lane >> 4;
  const int nbase = blockIdx.x * 64;
  const int mbase = blockIdx.y * 64;

  v8f acc0 = vzero8(), acc1 = vzero8(), acc2 = vzero8(), acc3 = vzero8();

  const float*          Af = (const float*)Aptr;
  const unsigned short* Ab = (const unsigned short*)Aptr;

  for (int k0 = 0; k0 < K; k0 += 32) {
#if USE_TDM
    // async: TDM pulls the 32x64 fp32 B tile into LDS while we stage A
    if (wave == 0) {
      tdm_load_2d(lds_off_u32(lBf), B + (size_t)k0 * N + nbase,
                  /*4B*/ 2, /*tile*/ 64, 32, /*tensor*/ (unsigned)N,
                  (unsigned)K, (unsigned long long)N);
    }
#endif
    // stage A tile 64x32 (convert fp32 -> bf16 if needed)
    {
      const int row  = tid >> 1;
      const int kp   = (tid & 1) * 16;
      const int grow = mbase + row;
      if (ABF16) {
        const unsigned short* src = Ab + (size_t)grow * K + k0 + kp;
#pragma unroll
        for (int i = 0; i < 16; ++i) lA[row * 32 + kp + i] = src[i];
      } else {
        const float* src = Af + (size_t)grow * K + k0 + kp;
#pragma unroll
        for (int i = 0; i < 16; ++i) lA[row * 32 + kp + i] = f2bf(src[i]);
      }
    }
#if USE_TDM
    if (wave == 0) __builtin_amdgcn_s_wait_tensorcnt(0);
    __syncthreads();
    // convert fp32 tile -> bf16, transposed into [n][k]
    for (int idx = tid; idx < 2048; idx += 128) {
      const int kk = idx >> 6;
      const int nn = idx & 63;
      lBt[nn * 32 + kk] = f2bf(lBf[kk * 64 + nn]);
    }
#else
    for (int idx = tid; idx < 2048; idx += 128) {
      const int kk = idx >> 6;
      const int nn = idx & 63;
      lBt[nn * 32 + kk] = f2bf(B[(size_t)(k0 + kk) * N + nbase + nn]);
    }
#endif
    __syncthreads();

    const int arow = wave * 16 + l16;
    const v16bf af = load_frag_2x8(&lA[arow * 32 + g * 8],
                                   &lA[arow * 32 + g * 8 + 16]);
#pragma unroll
    for (int t = 0; t < 4; ++t) {
      const v16bf bfr = load_frag_2x8(&lBt[(t * 16 + l16) * 32 + g * 8],
                                      &lBt[(t * 16 + l16) * 32 + g * 8 + 16]);
      v8f& acc = (t == 0 ? acc0 : t == 1 ? acc1 : t == 2 ? acc2 : acc3);
      acc = __builtin_amdgcn_wmma_f32_16x16x32_bf16(false, af, false, bfr,
                                                    (short)0, acc, false, false);
    }
    __syncthreads();
  }

#pragma unroll
  for (int t = 0; t < 4; ++t) {
    const v8f& acc = (t == 0 ? acc0 : t == 1 ? acc1 : t == 2 ? acc2 : acc3);
#pragma unroll
    for (int r = 0; r < 8; ++r) {
      const int row = mbase + wave * 16 + g * 8 + r;
      const int col = nbase + t * 16 + l16;
      C[(size_t)row * N + col] = acc[r];
    }
  }
}

// ---------------------------------------------------------------------------
// RoPE (rotate-half) + split xqkv fp32 -> Q/K/V bf16
// ---------------------------------------------------------------------------
__global__ __launch_bounds__(256) void rope_split_kernel(
    const float* __restrict__ xqkv, unsigned short* __restrict__ Qb,
    unsigned short* __restrict__ Kb, unsigned short* __restrict__ Vb) {
  const int idx = blockIdx.x * 256 + threadIdx.x;
  const int total = SLEN * (NHEADS + 2 * NKVH) * HEADD;
  if (idx >= total) return;
  const int d    = idx & (HEADD - 1);
  const int head = (idx >> 7) % (NHEADS + 2 * NKVH);
  const int s    = idx / ((NHEADS + 2 * NKVH) * HEADD);
  const float* x = xqkv + (size_t)s * QKV_N + head * HEADD;

  if (head >= NHEADS + NKVH) {  // V: plain copy
    Vb[((size_t)s * NKVH + (head - NHEADS - NKVH)) * HEADD + d] = f2bf(x[d]);
    return;
  }
  const int j = d & 63;
  const float inv = __powf(10000.0f, -(float)(2 * j) * (1.0f / 128.0f));
  const float ang = (float)s * inv;
  const float c = __cosf(ang), sn = __sinf(ang);
  const float a = x[j], b = x[j + 64];
  const float val = (d < 64) ? (a * c - b * sn) : (a * sn + b * c);
  if (head < NHEADS)
    Qb[((size_t)s * NHEADS + head) * HEADD + d] = f2bf(val);
  else
    Kb[((size_t)s * NKVH + (head - NHEADS)) * HEADD + d] = f2bf(val);
}

// ---------------------------------------------------------------------------
// Causal flash attention (GQA), one wave per (16-query tile, head).
// TDM streams the next V tile into LDS while Q.K^T WMMAs + softmax run.
// ---------------------------------------------------------------------------
__global__ __launch_bounds__(32) void flash_attn_kernel(
    const unsigned short* __restrict__ Qb, const unsigned short* __restrict__ Kb,
    const unsigned short* __restrict__ Vb, unsigned short* __restrict__ Ob) {
  __shared__ alignas(32) unsigned short lP[16 * 32];   // P tile [row][keylocal]
  __shared__ alignas(32) unsigned short lV[32 * 128];  // V tile [key][d]

  const int lane = threadIdx.x & 31;
  const int l16  = lane & 15;
  const int g    = lane >> 4;
  const int qb   = blockIdx.x;   // 16-query tile
  const int h    = blockIdx.y;   // head
  const int kvh  = h >> 2;       // GROUPS = 4

  // Q fragments (4 K-chunks of head_dim=128) kept in registers
  v16bf qf[4];
  {
    const unsigned short* qrow =
        Qb + ((size_t)(qb * 16 + l16) * NHEADS + h) * HEADD;
#pragma unroll
    for (int c = 0; c < 4; ++c) {
      const int d0 = c * 32 + g * 16;
      qf[c] = *(const v16bf*)(qrow + d0);
    }
  }

  v8f acc[8];
#pragma unroll
  for (int i = 0; i < 8; ++i) acc[i] = vzero8();
  float m[8], l[8];
#pragma unroll
  for (int r = 0; r < 8; ++r) { m[r] = -1e30f; l[r] = 0.f; }

  const int   jmax  = (qb * 16 + 15) >> 5;
  const float scale = 0.0883883476483184f;   // 1/sqrt(128)

  for (int j = 0; j <= jmax; ++j) {
    const int k0 = j << 5;

#if USE_TDM
    // async: V tile (32 keys x 128 d, row stride NKVH*HEADD) -> LDS
    tdm_load_2d(lds_off_u32(lV),
                Vb + ((size_t)k0 * NKVH + kvh) * HEADD,
                /*2B*/ 1, /*tile*/ HEADD, 32, /*tensor*/ HEADD,
                (unsigned)(SLEN * NKVH), (unsigned long long)(NKVH * HEADD));
#else
    for (int idx = lane; idx < 32 * HEADD; idx += 32) {
      const int key = idx >> 7;
      const int d   = idx & (HEADD - 1);
      lV[key * HEADD + d] = Vb[((size_t)(k0 + key) * NKVH + kvh) * HEADD + d];
    }
#endif

    // scores: two 16x16 tiles (keys k0..k0+15, k0+16..k0+31); overlaps the DMA
    v8f s0 = vzero8(), s1 = vzero8();
    {
      const unsigned short* kr0 =
          Kb + ((size_t)(k0 + l16) * NKVH + kvh) * HEADD;
      const unsigned short* kr1 =
          Kb + ((size_t)(k0 + 16 + l16) * NKVH + kvh) * HEADD;
#pragma unroll
      for (int c = 0; c < 4; ++c) {
        const int d0 = c * 32 + g * 16;
        const v16bf kf0 = *(const v16bf*)(kr0 + d0);
        const v16bf kf1 = *(const v16bf*)(kr1 + d0);
        s0 = __builtin_amdgcn_wmma_f32_16x16x32_bf16(false, qf[c], false, kf0,
                                                     (short)0, s0, false, false);
        s1 = __builtin_amdgcn_wmma_f32_16x16x32_bf16(false, qf[c], false, kf1,
                                                     (short)0, s1, false, false);
      }
    }

    // online softmax (rows live in 16-lane halves of the wave32)
    float alpha[8];
    const int key0 = k0 + l16, key1 = key0 + 16;
#pragma unroll
    for (int r = 0; r < 8; ++r) {
      const int q = qb * 16 + g * 8 + r;
      float a0 = s0[r] * scale; if (key0 > q) a0 = -1e30f;
      float a1 = s1[r] * scale; if (key1 > q) a1 = -1e30f;
      float mx = fmaxf(a0, a1);
#pragma unroll
      for (int off = 1; off < 16; off <<= 1)
        mx = fmaxf(mx, __shfl_xor(mx, off, 32));
      const float mn = fmaxf(m[r], mx);
      const float p0 = __expf(a0 - mn);
      const float p1 = __expf(a1 - mn);
      float rs = p0 + p1;
#pragma unroll
      for (int off = 1; off < 16; off <<= 1)
        rs += __shfl_xor(rs, off, 32);
      alpha[r] = __expf(m[r] - mn);
      l[r] = l[r] * alpha[r] + rs;
      m[r] = mn;
      lP[(g * 8 + r) * 32 + l16]      = f2bf(p0);
      lP[(g * 8 + r) * 32 + 16 + l16] = f2bf(p1);
    }

#if USE_TDM
    __builtin_amdgcn_s_wait_tensorcnt(0);
#endif
    __syncthreads();

    // rescale running accumulator
#pragma unroll
    for (int db = 0; db < 8; ++db)
#pragma unroll
      for (int r = 0; r < 8; ++r) acc[db][r] *= alpha[r];

    // P (A-layout) fragment from LDS; V (B-layout) gathered from [key][d]
    const v16bf pf = *(const v16bf*)&lP[l16 * 32 + g * 16];
#pragma unroll
    for (int db = 0; db < 8; ++db) {
      AFragU vf;
#pragma unroll
      for (int e = 0; e < 16; ++e)
        vf.u[e] = lV[(g * 16 + e) * HEADD + db * 16 + l16];
      acc[db] = __builtin_amdgcn_wmma_f32_16x16x32_bf16(
          false, pf, false, vf.v, (short)0, acc[db], false, false);
    }
    __syncthreads();
  }

  // normalize and store attention output as bf16 [s][h*128+d]
  float invl[8];
#pragma unroll
  for (int r = 0; r < 8; ++r) invl[r] = 1.0f / l[r];
#pragma unroll
  for (int db = 0; db < 8; ++db)
#pragma unroll
    for (int r = 0; r < 8; ++r) {
      const int row = qb * 16 + g * 8 + r;
      const int col = h * HEADD + db * 16 + l16;
      Ob[(size_t)row * (NHEADS * HEADD) + col] = f2bf(acc[db][r] * invl[r]);
    }
}

// ---------------------------------------------------------------------------
extern "C" void kernel_launch(void* const* d_in, const int* in_sizes, int n_in,
                              void* d_out, int out_size, void* d_ws,
                              size_t ws_size, hipStream_t stream) {
  (void)in_sizes; (void)n_in; (void)out_size; (void)ws_size;
  const float* hidden = (const float*)d_in[0];   // [1,2048,4096]
  const float* w_qkv  = (const float*)d_in[1];   // [4096,6144]
  const float* w_o    = (const float*)d_in[2];   // [4096,4096]
  // d_in[3] = attention_mask: exactly causal tril; applied analytically.

  float*          xqkv = (float*)d_ws;                            // 48 MB
  unsigned short* Qb = (unsigned short*)(xqkv + (size_t)SLEN * QKV_N);
  unsigned short* Kb = Qb + (size_t)SLEN * NHEADS * HEADD;
  unsigned short* Vb = Kb + (size_t)SLEN * NKVH * HEADD;
  unsigned short* Ob = Vb + (size_t)SLEN * NKVH * HEADD;

  // 1) QKV projection
  gemm_bf16_kernel<false><<<dim3(QKV_N / 64, SLEN / 64), 128, 0, stream>>>(
      hidden, w_qkv, xqkv, SLEN, QKV_N, HDIM);
  // 2) RoPE + split into bf16 Q/K/V
  {
    const int total = SLEN * (NHEADS + 2 * NKVH) * HEADD;
    rope_split_kernel<<<(total + 255) / 256, 256, 0, stream>>>(xqkv, Qb, Kb, Vb);
  }
  // 3) causal flash attention
  flash_attn_kernel<<<dim3(SLEN / 16, NHEADS), 32, 0, stream>>>(Qb, Kb, Vb, Ob);
  // 4) output projection -> fp32 d_out
  gemm_bf16_kernel<true><<<dim3(HDIM / 64, SLEN / 64), 128, 0, stream>>>(
      Ob, w_o, (float*)d_out, SLEN, HDIM, HDIM);
}
